// HardestContrastiveLoss_76381698392659
// MI455X (gfx1250) — compile-verified, matrix operand order checked by reading, not run on hardware
//
#include <hip/hip_runtime.h>

#define S_CNT 4096
#define C_CNT 32768
#define D_DIM 256
#define P_CNT 10000

typedef __attribute__((ext_vector_type(16))) __bf16 v16bf;
typedef __attribute__((ext_vector_type(8)))  float  v8f;

static __device__ __forceinline__ v8f wmma_bf16(v16bf a, v16bf b, v8f c) {
    // D = A(16x32 bf16) * B(32x16 bf16) + C(16x16 f32)
    return __builtin_amdgcn_wmma_f32_16x16x32_bf16(false, a, false, b, (short)0, c,
                                                   false, false);
}

// Fragment element order (ISA 7.12.2, 16-bit A/B 16x32):
//   k (0..31 within chunk) -> half h = (k>>3)&1, element e = (k&7) + ((k&16)>>1)
//   stored per row at [chunk*32 + h*16 + e], so a lane's 16 values are one
//   contiguous 32-byte run at row + chunk*32 + half*16.

// ---- prep: gather candidate rows, split f32 -> bf16 hi/lo, swizzle, norms ----
__global__ __launch_bounds__(256) void hcl_pack_cand_kernel(
    const float* __restrict__ feats0, const float* __restrict__ feats1,
    const int* __restrict__ ci0, const int* __restrict__ ci1,
    __bf16* __restrict__ chi, __bf16* __restrict__ clo, float* __restrict__ cnorm)
{
    const int gw   = (blockIdx.x * 256 + threadIdx.x) >> 5;  // global wave = row slot
    const int lane = threadIdx.x & 31;
    if (gw >= 2 * C_CNT) return;
    const int dir = gw >> 15;
    const int col = gw & (C_CNT - 1);

    const float* f  = dir ? feats0 : feats1;
    const int*   ci = dir ? ci0 : ci1;
    const float* src = f + (size_t)ci[col] * D_DIM;

    __bf16* dh = chi + (size_t)gw * D_DIM;
    __bf16* dl = clo + (size_t)gw * D_DIM;

    const int h = (lane >> 3) & 1;
    const int e = (lane & 7) + ((lane & 16) >> 1);
    float nrm = 0.0f;
#pragma unroll
    for (int c = 0; c < 8; ++c) {
        const float x = src[c * 32 + lane];
        nrm += x * x;
        const __bf16 hb = (__bf16)x;
        dh[c * 32 + h * 16 + e] = hb;
        dl[c * 32 + h * 16 + e] = (__bf16)(x - (float)hb);
    }
#pragma unroll
    for (int m = 16; m >= 1; m >>= 1) nrm += __shfl_xor(nrm, m, 32);
    if (lane == 0) cnorm[gw] = nrm;
}

// ---- prep: same for the sampled positive (query) rows ----
__global__ __launch_bounds__(256) void hcl_pack_query_kernel(
    const float* __restrict__ feats0, const float* __restrict__ feats1,
    const int* __restrict__ pp, const int* __restrict__ ppi,
    __bf16* __restrict__ qhi, __bf16* __restrict__ qlo, float* __restrict__ qnorm)
{
    const int gw   = (blockIdx.x * 256 + threadIdx.x) >> 5;
    const int lane = threadIdx.x & 31;
    if (gw >= 2 * S_CNT) return;
    const int dir = gw >> 12;
    const int i   = gw & (S_CNT - 1);

    const int sidx = ppi[i];
    const int qid  = pp[2 * sidx + dir];
    const float* src = (dir ? feats1 : feats0) + (size_t)qid * D_DIM;

    __bf16* dh = qhi + (size_t)gw * D_DIM;
    __bf16* dl = qlo + (size_t)gw * D_DIM;

    const int h = (lane >> 3) & 1;
    const int e = (lane & 7) + ((lane & 16) >> 1);
    float nrm = 0.0f;
#pragma unroll
    for (int c = 0; c < 8; ++c) {
        const float x = src[c * 32 + lane];
        nrm += x * x;
        const __bf16 hb = (__bf16)x;
        dh[c * 32 + h * 16 + e] = hb;
        dl[c * 32 + h * 16 + e] = (__bf16)(x - (float)hb);
    }
#pragma unroll
    for (int m = 16; m >= 1; m >>= 1) nrm += __shfl_xor(nrm, m, 32);
    if (lane == 0) qnorm[gw] = nrm;
}

// ---- main: WMMA distance sweep + per-row min/argmin ----
__global__ __launch_bounds__(256) void hcl_gemm_min_kernel(
    const __bf16* __restrict__ qhi, const __bf16* __restrict__ qlo,
    const __bf16* __restrict__ chi, const __bf16* __restrict__ clo,
    const float* __restrict__ qnorm, const float* __restrict__ cnorm,
    const int* __restrict__ ci0, const int* __restrict__ ci1,
    float* __restrict__ nd_out, int* __restrict__ nj_out)
{
    const int dir   = blockIdx.y;
    const int qbase = blockIdx.x * 16;
    const int wave  = threadIdx.x >> 5;
    const int lane  = threadIdx.x & 31;
    const int half  = lane >> 4;
    const int row   = lane & 15;

    // A fragments: resident in registers for the whole candidate sweep
    const size_t arow = (size_t)(dir * S_CNT + qbase + row) * D_DIM + half * 16;
    v16bf ahi[8], alo[8];
#pragma unroll
    for (int c = 0; c < 8; ++c) {
        ahi[c] = *(const v16bf*)(qhi + arow + c * 32);
        alo[c] = *(const v16bf*)(qlo + arow + c * 32);
    }
    float qn[8];
#pragma unroll
    for (int r = 0; r < 8; ++r)
        qn[r] = qnorm[dir * S_CNT + qbase + r + 8 * half];

    float bestv[8];
    int   bestj[8];
#pragma unroll
    for (int r = 0; r < 8; ++r) { bestv[r] = 3.4e38f; bestj[r] = 0; }

    for (int nb = wave * 16; nb < C_CNT; nb += 128) {
        const int ncol = nb + row;
        const size_t cbase = (size_t)(dir * C_CNT + ncol) * D_DIM + half * 16;
        const __bf16* bhp = chi + cbase;
        const __bf16* blp = clo + cbase;

        v8f a0 = {}, a1 = {}, a2 = {};   // 3 independent accumulation chains
#pragma unroll
        for (int c = 0; c < 8; ++c) {
            const v16bf bh = *(const v16bf*)(bhp + c * 32);
            const v16bf bl = *(const v16bf*)(blp + c * 32);
            a0 = wmma_bf16(ahi[c], bh, a0);   // hi*hi
            a1 = wmma_bf16(ahi[c], bl, a1);   // hi*lo
            a2 = wmma_bf16(alo[c], bh, a2);   // lo*hi
        }
        const float cn = cnorm[dir * C_CNT + ncol];
#pragma unroll
        for (int r = 0; r < 8; ++r) {
            const float d = qn[r] + cn - 2.0f * (a0[r] + a1[r] + a2[r]);
            if (d < bestv[r]) { bestv[r] = d; bestj[r] = ncol; }  // keeps lowest j on ties
        }
    }

    // reduce min/argmin over the 16 lanes of each half (same row set)
#pragma unroll
    for (int r = 0; r < 8; ++r) {
        float v = bestv[r];
        int   j = bestj[r];
#pragma unroll
        for (int m = 1; m <= 8; m <<= 1) {
            float ov = __shfl_xor(v, m, 32);
            int   oj = __shfl_xor(j, m, 32);
            if (ov < v || (ov == v && oj < j)) { v = ov; j = oj; }
        }
        bestv[r] = v;
        bestj[r] = j;
    }

    __shared__ float red_v[8][16];
    __shared__ int   red_j[8][16];
    if (row == 0) {
#pragma unroll
        for (int r = 0; r < 8; ++r) {
            red_v[wave][r + 8 * half] = bestv[r];
            red_j[wave][r + 8 * half] = bestj[r];
        }
    }
    __syncthreads();

    if (threadIdx.x < 16) {
        const int m = threadIdx.x;
        float v = red_v[0][m];
        int   j = red_j[0][m];
#pragma unroll
        for (int w = 1; w < 8; ++w) {
            float ov = red_v[w][m];
            int   oj = red_j[w][m];
            if (ov < v || (ov == v && oj < j)) { v = ov; j = oj; }
        }
        const int* cidx = dir ? ci0 : ci1;
        nd_out[dir * S_CNT + qbase + m] = v;
        nj_out[dir * S_CNT + qbase + m] = cidx[j];
    }
}

__global__ __launch_bounds__(256) void hcl_pass1_kernel(
    const float* __restrict__ feats0, const float* __restrict__ feats1,
    const int* __restrict__ pp, const int* __restrict__ ppi,
    const float* __restrict__ nd, const int* __restrict__ nj,
    float* __restrict__ posv, float* __restrict__ negv, float* __restrict__ negc)
{
    const int t = blockIdx.x * 256 + threadIdx.x;
    if (t >= 2 * S_CNT) return;
    const int dir = t >> 12;
    const int i   = t & (S_CNT - 1);

    const int sidx = ppi[i];
    const int a = pp[2 * sidx + 0];
    const int b = pp[2 * sidx + 1];
    const int qid = dir ? b : a;
    const int ni  = nj[dir * S_CNT + i];

    const int c0 = dir ? ni  : qid;
    const int c1 = dir ? qid : ni;
    bool hit = false;
    for (int j = 0; j < P_CNT; ++j) {
        hit = hit || (pp[2 * j] == c0 && pp[2 * j + 1] == c1);
    }

    const float v  = nd[dir * S_CNT + i];
    const float rl = fmaxf(1.4f - v, 0.0f);
    negv[t] = hit ? 0.0f : rl * rl;
    negc[t] = hit ? 0.0f : 1.0f;

    if (dir == 0) {
        const float* r0 = feats0 + (size_t)a * D_DIM;
        const float* r1 = feats1 + (size_t)b * D_DIM;
        float s = 0.0f;
        for (int k = 0; k < D_DIM; k += 4) {
            float4 x = *(const float4*)(r0 + k);
            float4 y = *(const float4*)(r1 + k);
            float dx = x.x - y.x, dy = x.y - y.y, dz = x.z - y.z, dw = x.w - y.w;
            s += dx * dx + dy * dy + dz * dz + dw * dw;
        }
        posv[i] = fmaxf(s - 0.1f, 0.0f);
    }
}

__global__ __launch_bounds__(256) void hcl_pass2_kernel(
    const float* __restrict__ posv, const float* __restrict__ negv,
    const float* __restrict__ negc, float* __restrict__ out)
{
    __shared__ float sh[5][256];
    float ap = 0.f, v0 = 0.f, c0 = 0.f, v1 = 0.f, c1 = 0.f;
    for (int i = threadIdx.x; i < S_CNT; i += 256) {
        ap += posv[i];
        v0 += negv[i];          c0 += negc[i];
        v1 += negv[S_CNT + i];  c1 += negc[S_CNT + i];
    }
    sh[0][threadIdx.x] = ap;
    sh[1][threadIdx.x] = v0;
    sh[2][threadIdx.x] = c0;
    sh[3][threadIdx.x] = v1;
    sh[4][threadIdx.x] = c1;
    __syncthreads();
    for (int s = 128; s > 0; s >>= 1) {
        if (threadIdx.x < s) {
#pragma unroll
            for (int q = 0; q < 5; ++q)
                sh[q][threadIdx.x] += sh[q][threadIdx.x + s];
        }
        __syncthreads();
    }
    if (threadIdx.x == 0) {
        const float pos = sh[0][0] / (float)S_CNT;
        const float nl0 = sh[1][0] / fmaxf(sh[2][0], 1.0f);
        const float nl1 = sh[3][0] / fmaxf(sh[4][0], 1.0f);
        const float neg = 0.5f * (nl0 + nl1);
        out[0] = pos + neg;
        out[1] = pos;
        out[2] = neg;
    }
}

extern "C" void kernel_launch(void* const* d_in, const int* in_sizes, int n_in,
                              void* d_out, int out_size, void* d_ws, size_t ws_size,
                              hipStream_t stream) {
    const float* feats0 = (const float*)d_in[0];
    const float* feats1 = (const float*)d_in[1];
    const int*   pp     = (const int*)d_in[2];
    const int*   ci0    = (const int*)d_in[3];
    const int*   ci1    = (const int*)d_in[4];
    const int*   ppi    = (const int*)d_in[5];
    float* out = (float*)d_out;

    // workspace carve-up (~72.5 MB)
    __bf16* chi = (__bf16*)d_ws;                         // 2*C*D bf16 = 32 MB
    __bf16* clo = chi + (size_t)2 * C_CNT * D_DIM;       // 32 MB
    __bf16* qhi = clo + (size_t)2 * C_CNT * D_DIM;       // 4 MB
    __bf16* qlo = qhi + (size_t)2 * S_CNT * D_DIM;       // 4 MB
    float* cnorm = (float*)(qlo + (size_t)2 * S_CNT * D_DIM);
    float* qnorm = cnorm + 2 * C_CNT;
    float* nd    = qnorm + 2 * S_CNT;
    int*   nj    = (int*)(nd + 2 * S_CNT);
    float* negv  = (float*)(nj + 2 * S_CNT);
    float* negc  = negv + 2 * S_CNT;
    float* posv  = negc + 2 * S_CNT;

    hcl_pack_cand_kernel<<<(2 * C_CNT) / 8, 256, 0, stream>>>(
        feats0, feats1, ci0, ci1, chi, clo, cnorm);
    hcl_pack_query_kernel<<<(2 * S_CNT) / 8, 256, 0, stream>>>(
        feats0, feats1, pp, ppi, qhi, qlo, qnorm);

    dim3 g1(S_CNT / 16, 2);
    hcl_gemm_min_kernel<<<g1, 256, 0, stream>>>(qhi, qlo, chi, clo, qnorm, cnorm,
                                                ci0, ci1, nd, nj);
    hcl_pass1_kernel<<<(2 * S_CNT) / 256, 256, 0, stream>>>(feats0, feats1, pp, ppi,
                                                            nd, nj, posv, negv, negc);
    hcl_pass2_kernel<<<1, 256, 0, stream>>>(posv, negv, negc, out);
}
